// HEDNet_91242285236695
// MI455X (gfx1250) — matrix operand order repaired and checked by reading, not compiled, not run on hardware
//
#include <hip/hip_runtime.h>
#include <hip/hip_bf16.h>
#include <math.h>

// ---------------------------------------------------------------------------
// Types for CDNA5 WMMA (gfx1250, wave32)
// ---------------------------------------------------------------------------
typedef __attribute__((ext_vector_type(16))) __bf16 v16bf;
typedef __attribute__((ext_vector_type(8)))  __bf16 v8bf;
typedef __attribute__((ext_vector_type(4)))  __bf16 v4bf;
typedef __attribute__((ext_vector_type(8)))  float  v8f;
typedef __attribute__((ext_vector_type(4)))  int    v4i;
typedef __attribute__((ext_vector_type(2)))  int    v2i;

#define BN_EPS 1e-3f

// gfx1250 async memory->LDS path (ASYNCcnt-tracked), probed via __has_builtin.
#if defined(__has_builtin)
#if __has_builtin(__builtin_amdgcn_global_load_async_to_lds_b128) && \
    __has_builtin(__builtin_amdgcn_global_load_async_to_lds_b64)
#define HAVE_ASYNC_LDS 1
#endif
#endif
#ifndef HAVE_ASYNC_LDS
#define HAVE_ASYNC_LDS 0
#endif

#define GAS __attribute__((address_space(1)))
#define LAS __attribute__((address_space(3)))

__device__ __forceinline__ void async_wait_then_barrier() {
#if HAVE_ASYNC_LDS
#if __has_builtin(__builtin_amdgcn_s_wait_asynccnt)
  __builtin_amdgcn_s_wait_asynccnt(0);
#else
  asm volatile("s_wait_asynccnt 0x0" ::: "memory");
#endif
#endif
  __syncthreads();
}

__device__ __forceinline__ __bf16 f2bf(float f) {
  // round-to-nearest-even f32 -> bf16
  unsigned u = __builtin_bit_cast(unsigned, f);
  unsigned r = (u + 0x7FFFu + ((u >> 16) & 1u)) >> 16;
  unsigned short s = (unsigned short)r;
  return __builtin_bit_cast(__bf16, s);
}

// ---------------------------------------------------------------------------
// Weight repack: f32 OIDHW -> bf16 [Opad][Kpad], k = tap*Cin + ci, zero pad.
// Runs once per conv layer per launch; makes B staging an unconditional,
// fully-coalesced contiguous copy (async DMA to LDS when available).
// ---------------------------------------------------------------------------
__global__ void repack_w_kernel(const float* __restrict__ w,
                                __bf16* __restrict__ o_,
                                int Cout, int Cin, int taps, int Kpad, int Opad)
{
  int i = blockIdx.x * blockDim.x + threadIdx.x;
  int total = Opad * Kpad;
  if (i >= total) return;
  int o = i / Kpad, k = i - o * Kpad;
  float v = 0.f;
  if (o < Cout && k < taps * Cin) {
    int tap = k / Cin; int ci = k - tap * Cin;
    v = w[((size_t)o * Cin + ci) * taps + tap];
  }
  o_[i] = f2bf(v);
}

// ---------------------------------------------------------------------------
// Implicit-GEMM 3D conv, double-buffered LDS, fused bias/BN/res/ReLU/mask.
// Activations NDHWC; weights pre-repacked bf16 [Opad][Kpad].
// GEMM: M = out voxels, N = Cout, K = taps*Cin (k = tap*Cin + ci).
//
//   WN : N-wave split. WN=2 -> block tile 128x64 (4x2 waves)
//                      WN=1 -> block tile 256x32 (8x1 waves)
//        each wave computes 32x32 via 2x2 v_wmma_f32_16x16x32_bf16.
//   LD : lhs dilation (1 = normal conv, 2 = transposed conv gather)
//   L  : log2(Cin) for Cin in {16,32,64,128}; -1 = generic (stem Cin=5)
// One workgroup barrier per 32-K step (stage k+1 into buf^1, WMMA from buf).
// ---------------------------------------------------------------------------
struct ConvParams {
  int Din, Hin, Win, Cin;
  int Dout, Hout, Wout, Cout;
  int kd, kh, kw;
  int sz, sy, sx;
  int pz, py, px;
  int Kpad;
  int mode;  // 0: bn only   1: relu(bn(y))*mask   2: relu(bn(y)+res)*mask
};

template <int WN, int LD, int L>
__global__ __launch_bounds__(256) void conv_wmma(
    const float* __restrict__ in, const __bf16* __restrict__ wbf,
    const float* __restrict__ bn_b, const float* __restrict__ bn_g,
    const float* __restrict__ bn_m, const float* __restrict__ bn_v,
    const float* __restrict__ bias, const float* __restrict__ res,
    const float* __restrict__ mask, float* __restrict__ out, ConvParams P)
{
  constexpr int NWM   = 8 / WN;        // waves along M
  constexpr int BM    = NWM * 32;      // 256 (WN1) or 128 (WN2)
  constexpr int BN    = WN * 32;       // 32 or 64
  constexpr int LDA   = 40;            // padded LDS row stride (bank spread)
  constexpr int APASS = BM / 32;
  constexpr int TPR   = 256 / BN;      // threads per B row
  constexpr int BCOLS = 32 / TPR;      // 4 (WN1) or 8 (WN2)

  __shared__ __align__(16) __bf16 As[2][BM * LDA];
  __shared__ __align__(16) __bf16 Bs[2][BN * LDA];

  const int tid  = threadIdx.x;
  const int lane = tid & 31;
  const int wv   = tid >> 5;
  const int wM   = wv / WN;
  const int wNi  = wv % WN;

  const int M    = P.Dout * P.Hout * P.Wout;
  const int HWo  = P.Hout * P.Wout;
  const int m0   = blockIdx.x * BM;
  const int n0   = blockIdx.y * BN;

  const int taps = P.kd * P.kh * P.kw;
  const int khw  = P.kh * P.kw;
  const int Cin  = P.Cin;
  const int Ktot = taps * Cin;
  const int niter = P.Kpad >> 5;

  const int a_row = tid >> 3;              // 0..31
  const int a_col = (tid & 7) << 2;        // 0..28 step 4
  const int b_row = tid / TPR;             // 0..BN-1
  const int b_col = (tid % TPR) * BCOLS;

  // ---- hoisted per-pass voxel decode (loop invariant) ----
  bool mok[APASS];
  int  zb[APASS], yb[APASS], xb[APASS];
  #pragma unroll
  for (int p = 0; p < APASS; ++p) {
    int m = m0 + p * 32 + a_row;
    mok[p] = (m < M);
    int mm  = mok[p] ? m : 0;
    int zo  = mm / HWo;  int rem = mm - zo * HWo;
    int yo  = rem / P.Wout; int xo = rem - yo * P.Wout;
    zb[p] = zo * P.sz - P.pz;
    yb[p] = yo * P.sy - P.py;
    xb[p] = xo * P.sx - P.px;
  }

  // ---- incremental tap iterator for A (flat-K path) ----
  int A_tap = 0, A_td = 0, A_th = 0, A_tw = 0;
  if constexpr (L >= 0) {
    A_tap = a_col >> L;
    A_td = A_tap / khw; { int r = A_tap - A_td * khw; A_th = r / P.kw; A_tw = r - A_th * P.kw; }
  }
  auto advanceA = [&](int it) {          // prepare iterator for stage it+1
    if constexpr (L >= 0) {
      int nt = (((it + 1) << 5) + a_col) >> L;
      while (A_tap < nt) {
        ++A_tap;
        if (++A_tw == P.kw) { A_tw = 0; if (++A_th == P.kh) { A_th = 0; ++A_td; } }
      }
    }
  };

  auto stage = [&](int it, int buf) {
    const int k0 = it << 5;
    // ---------------- B tile: unconditional contiguous copy ----------------
    {
      const __bf16* src = wbf + (size_t)(n0 + b_row) * P.Kpad + k0 + b_col;
      __bf16* dst = &Bs[buf][b_row * LDA + b_col];
#if HAVE_ASYNC_LDS
      if constexpr (BCOLS == 8)
        __builtin_amdgcn_global_load_async_to_lds_b128(
            (GAS v4i*)src, (LAS v4i*)dst, 0, 0);
      else
        __builtin_amdgcn_global_load_async_to_lds_b64(
            (GAS v2i*)src, (LAS v2i*)dst, 0, 0);
#else
      if constexpr (BCOLS == 8) *(v8bf*)dst = *(const v8bf*)src;
      else                      *(v4bf*)dst = *(const v4bf*)src;
#endif
    }
    // ---------------- A tile (f32 gather -> bf16 LDS) ----------------
    #pragma unroll
    for (int p = 0; p < APASS; ++p) {
      float v0 = 0.f, v1 = 0.f, v2 = 0.f, v3 = 0.f;
      if constexpr (L >= 0) {
        const int ci = (k0 + a_col) & (Cin - 1);
        int iz = zb[p] + A_td, iy = yb[p] + A_th, ix = xb[p] + A_tw;
        bool ok = mok[p] && (A_tap < taps) && (iz >= 0) && (iy >= 0) && (ix >= 0);
        if constexpr (LD == 2) {
          ok = ok && (((iz | iy | ix) & 1) == 0);
          iz >>= 1; iy >>= 1; ix >>= 1;
        }
        ok = ok && (iz < P.Din) && (iy < P.Hin) && (ix < P.Win);
        if (ok) {
          const float4 f4 = *(const float4*)(in +
              ((size_t)(iz * P.Hin + iy) * P.Win + ix) * Cin + ci);
          v0 = f4.x; v1 = f4.y; v2 = f4.z; v3 = f4.w;
        }
      } else {
        float vv[4] = {0.f, 0.f, 0.f, 0.f};
        #pragma unroll
        for (int e = 0; e < 4; ++e) {
          const int k = k0 + a_col + e;
          if (mok[p] && k < Ktot) {
            int tap = k / Cin;  int ci = k - tap * Cin;
            int td = tap / khw; int r = tap - td * khw;
            int th = r / P.kw;  int tw = r - th * P.kw;
            int iz = zb[p] + td, iy = yb[p] + th, ix = xb[p] + tw;
            bool ok = (iz >= 0) && (iy >= 0) && (ix >= 0);
            if constexpr (LD == 2) {
              ok = ok && (((iz | iy | ix) & 1) == 0);
              iz >>= 1; iy >>= 1; ix >>= 1;
            }
            ok = ok && (iz < P.Din) && (iy < P.Hin) && (ix < P.Win);
            if (ok) vv[e] = in[((size_t)(iz * P.Hin + iy) * P.Win + ix) * Cin + ci];
          }
        }
        v0 = vv[0]; v1 = vv[1]; v2 = vv[2]; v3 = vv[3];
      }
      v4bf pk; pk[0] = f2bf(v0); pk[1] = f2bf(v1); pk[2] = f2bf(v2); pk[3] = f2bf(v3);
      *(v4bf*)(&As[buf][(p * 32 + a_row) * LDA + a_col]) = pk;
    }
  };

  v8f acc[2][2] = {};
  const int half = lane >> 4;
  const int l15  = lane & 15;

  auto compute = [&](int buf) {
    v16bf afr[2], bfr[2];
    #pragma unroll
    for (int mt = 0; mt < 2; ++mt) {
      const int r = wM * 32 + mt * 16 + l15;
      union { v16bf v; v8bf h[2]; } u;
      u.h[0] = *(const v8bf*)(&As[buf][r * LDA + (half << 3)]);       // K 0..7 | 8..15
      u.h[1] = *(const v8bf*)(&As[buf][r * LDA + 16 + (half << 3)]);  // K 16..23 | 24..31
      afr[mt] = u.v;
    }
    #pragma unroll
    for (int nt = 0; nt < 2; ++nt) {
      const int n = wNi * 32 + nt * 16 + l15;
      union { v16bf v; v8bf h[2]; } u;
      u.h[0] = *(const v8bf*)(&Bs[buf][n * LDA + (half << 3)]);
      u.h[1] = *(const v8bf*)(&Bs[buf][n * LDA + 16 + (half << 3)]);
      bfr[nt] = u.v;
    }
    #pragma unroll
    for (int mt = 0; mt < 2; ++mt)
      #pragma unroll
      for (int nt = 0; nt < 2; ++nt)
        acc[mt][nt] = __builtin_amdgcn_wmma_f32_16x16x32_bf16(
            false, afr[mt], false, bfr[nt], (short)0, acc[mt][nt],
            false, false);
  };

  // ---- double-buffered main loop: one barrier per 32-K step ----
  stage(0, 0);
  advanceA(0);
  async_wait_then_barrier();
  for (int it = 0; it < niter; ++it) {
    const int cur = it & 1;
    if (it + 1 < niter) {
      stage(it + 1, cur ^ 1);
      advanceA(it + 1);
    }
    compute(cur);
    async_wait_then_barrier();
  }

  // ---- fused epilogue: bias folded into BN shift, residual, ReLU, mask ----
  const int mh = half << 3;
  #pragma unroll
  for (int nt = 0; nt < 2; ++nt) {
    const int n = n0 + wNi * 32 + nt * 16 + l15;
    if (n >= P.Cout) continue;
    const float sc = bn_g[n] * (1.0f / sqrtf(bn_v[n] + BN_EPS));
    float sh = bn_b[n] - bn_m[n] * sc;
    if (bias) sh += bias[n] * sc;
    #pragma unroll
    for (int mt = 0; mt < 2; ++mt) {
      const int mb = m0 + wM * 32 + mt * 16 + mh;
      #pragma unroll
      for (int r = 0; r < 8; ++r) {
        const int m = mb + r;
        if (m >= M) continue;
        float val = acc[mt][nt][r] * sc + sh;
        if (P.mode == 2) val += res[(size_t)m * P.Cout + n];
        if (P.mode >= 1) {
          val = fmaxf(val, 0.f);
          val *= mask[m];
        }
        out[(size_t)m * P.Cout + n] = val;
      }
    }
  }
}

// ---------------------------------------------------------------------------
// Helper kernels
// ---------------------------------------------------------------------------
__global__ void scatter_kernel(const float* __restrict__ vf,
                               const int* __restrict__ coors,
                               float* __restrict__ dense,
                               float* __restrict__ mask,
                               int N, int D, int H, int W, int C)
{
  int i = blockIdx.x * blockDim.x + threadIdx.x;
  if (i >= N) return;
  int z = coors[4 * i + 1], y = coors[4 * i + 2], x = coors[4 * i + 3];
  if (z < 0 || z >= D || y < 0 || y >= H || x < 0 || x >= W) return;
  size_t m = ((size_t)z * H + y) * W + x;
  for (int c = 0; c < C; ++c) dense[m * C + c] = vf[(size_t)i * C + c];
  mask[m] = 1.f;
}

__global__ void dilate_mask_kernel(const float* __restrict__ mi,
                                   float* __restrict__ mo,
                                   int Din, int Hin, int Win,
                                   int Dout, int Hout, int Wout,
                                   int kd, int kh, int kw,
                                   int sz, int sy, int sx,
                                   int pz, int py, int px)
{
  int m = blockIdx.x * blockDim.x + threadIdx.x;
  int M = Dout * Hout * Wout;
  if (m >= M) return;
  int zo = m / (Hout * Wout);
  int rem = m - zo * (Hout * Wout);
  int yo = rem / Wout, xo = rem - (rem / Wout) * Wout;
  float s = 0.f;
  for (int a = 0; a < kd; ++a) {
    int iz = zo * sz - pz + a; if (iz < 0 || iz >= Din) continue;
    for (int b = 0; b < kh; ++b) {
      int iy = yo * sy - py + b; if (iy < 0 || iy >= Hin) continue;
      for (int c = 0; c < kw; ++c) {
        int ix = xo * sx - px + c; if (ix < 0 || ix >= Win) continue;
        s += mi[((size_t)iz * Hin + iy) * Win + ix];
      }
    }
  }
  mo[m] = (s > 0.f) ? 1.f : 0.f;
}

__global__ void bn_add_mask_kernel(const float* __restrict__ x,
                                   const float* __restrict__ tgt,
                                   const float* __restrict__ g,
                                   const float* __restrict__ b,
                                   const float* __restrict__ mn,
                                   const float* __restrict__ vr,
                                   const float* __restrict__ mask,
                                   float* __restrict__ out,
                                   long long total, int C)
{
  long long i = (long long)blockIdx.x * blockDim.x + threadIdx.x;
  if (i >= total) return;
  int n = (int)(i % C);
  long long m = i / C;
  float sc = g[n] * (1.0f / sqrtf(vr[n] + BN_EPS));
  float sh = b[n] - mn[n] * sc;
  out[i] = ((x[i] + tgt[i]) * sc + sh) * mask[m];
}

__global__ void to_ncdhw_kernel(const float* __restrict__ src,
                                float* __restrict__ dst,
                                int C, int D, int H, int W)
{
  int i = blockIdx.x * blockDim.x + threadIdx.x;
  int total = C * D * H * W;
  if (i >= total) return;
  int dhw = D * H * W;
  int c = i / dhw;
  int rem = i - c * dhw;
  dst[i] = src[(size_t)rem * C + c];
}

// ---------------------------------------------------------------------------
// Host-side orchestration
// ---------------------------------------------------------------------------
namespace {

struct Dims { int D, H, W; };
static inline size_t vox(Dims d) { return (size_t)d.D * d.H * d.W; }
static inline int cdim(int n, int k, int s, int p) { return (n + 2 * p - k) / s + 1; }

struct WS { char* base; size_t off, cap; };
static void* wsAllocBytes(WS& w, size_t bytes) {
  bytes = (bytes + 255) & ~(size_t)255;
  void* p = (void*)(w.base + w.off);
  w.off += bytes;
  return p;
}
static float* wsAlloc(WS& w, size_t nfloats) {
  return (float*)wsAllocBytes(w, nfloats * sizeof(float));
}

struct Cursor { void* const* in; int idx; };
static const float* nxt(Cursor& c) { return (const float*)c.in[c.idx++]; }

struct BNp { const float *b, *g, *m, *v; };
static BNp rdBN(Cursor& c) { BNp r; r.b = nxt(c); r.g = nxt(c); r.m = nxt(c); r.v = nxt(c); return r; }
struct PABp { BNp bn; const float* w; };
static PABp rdPAB(Cursor& c) { PABp p; p.bn = rdBN(c); p.w = nxt(c); return p; }
struct SBBp { const float *b1, *b2; BNp bn1, bn2; const float *w1, *w2; };
static SBBp rdSBB(Cursor& c) {
  SBBp s; s.b1 = nxt(c); s.b2 = nxt(c); s.bn1 = rdBN(c); s.bn2 = rdBN(c);
  s.w1 = nxt(c); s.w2 = nxt(c); return s;
}
struct DECp { BNp bn, norm; const float* w; };
struct SEDp { DECp dec[2]; PABp first[3]; int nsbb[3]; SBBp sbbs[3][2]; };
static SEDp rdSED(Cursor& c) {
  SEDp s;
  for (int i = 0; i < 2; ++i) { s.dec[i].bn = rdBN(c); s.dec[i].norm = rdBN(c); s.dec[i].w = nxt(c); }
  const int NS[3] = {2, 1, 1};
  for (int j = 0; j < 3; ++j) {
    if (j > 0) s.first[j] = rdPAB(c);            // level 0 has stride 1 -> no 'first'
    s.nsbb[j] = NS[j];
    for (int q = 0; q < NS[j]; ++q) s.sbbs[j][q] = rdSBB(c);
  }
  return s;
}

static void conv_launch(hipStream_t st, WS& ws, const float* in, Dims di, int Cin,
                        const float* w, int Cout,
                        int kd, int kh, int kw, int sz, int sy, int sx,
                        int pz, int py, int px, int ld,
                        BNp bn, const float* bias, const float* res,
                        const float* mask, int mode, float* out, Dims dout)
{
  const int taps = kd * kh * kw;
  const int Ktot = taps * Cin;
  const int Kpad = (Ktot + 31) & ~31;
  const int Opad = (Cout + 63) & ~63;

  // per-layer weight repack to bf16 [Opad][Kpad]
  __bf16* wbf = (__bf16*)wsAllocBytes(ws, (size_t)Opad * Kpad * sizeof(__bf16));
  {
    int total = Opad * Kpad;
    repack_w_kernel<<<(total + 255) / 256, 256, 0, st>>>(
        w, wbf, Cout, Cin, taps, Kpad, Opad);
  }

  ConvParams P;
  P.Din = di.D; P.Hin = di.H; P.Win = di.W; P.Cin = Cin;
  P.Dout = dout.D; P.Hout = dout.H; P.Wout = dout.W; P.Cout = Cout;
  P.kd = kd; P.kh = kh; P.kw = kw;
  P.sz = sz; P.sy = sy; P.sx = sx;
  P.pz = pz; P.py = py; P.px = px;
  P.Kpad = Kpad;
  P.mode = mode;
  const int M   = dout.D * dout.H * dout.W;
  const int BM  = (Cout >= 64) ? 128 : 256;
  const int BNt = (Cout >= 64) ? 64 : 32;
  dim3 grid((M + BM - 1) / BM, (Cout + BNt - 1) / BNt);

#define LAUNCH_CONV(WNv, LDv, Lv)                                          \
  conv_wmma<WNv, LDv, Lv><<<grid, 256, 0, st>>>(                           \
      in, wbf, bn.b, bn.g, bn.m, bn.v, bias, res, mask, out, P)

  if (Cout >= 64) {
    if (ld == 1) {
      if      (Cin == 128) LAUNCH_CONV(2, 1, 7);
      else if (Cin == 64)  LAUNCH_CONV(2, 1, 6);
      else if (Cin == 32)  LAUNCH_CONV(2, 1, 5);
      else                 LAUNCH_CONV(2, 1, -1);
    } else {
      if      (Cin == 128) LAUNCH_CONV(2, 2, 7);
      else if (Cin == 64)  LAUNCH_CONV(2, 2, 6);
      else                 LAUNCH_CONV(2, 2, -1);
    }
  } else {
    if (ld == 1) {
      if      (Cin == 32)  LAUNCH_CONV(1, 1, 5);
      else if (Cin == 16)  LAUNCH_CONV(1, 1, 4);
      else                 LAUNCH_CONV(1, 1, -1);
    } else {
      if      (Cin == 32)  LAUNCH_CONV(1, 2, 5);
      else if (Cin == 16)  LAUNCH_CONV(1, 2, 4);
      else                 LAUNCH_CONV(1, 2, -1);
    }
  }
#undef LAUNCH_CONV
}

static void dilate_launch(hipStream_t st, const float* mi, Dims di,
                          float* mo, Dims dout,
                          int kd, int kh, int kw, int sz, int sy, int sx,
                          int pz, int py, int px)
{
  int M = dout.D * dout.H * dout.W;
  dilate_mask_kernel<<<(M + 255) / 256, 256, 0, st>>>(
      mi, mo, di.D, di.H, di.W, dout.D, dout.H, dout.W,
      kd, kh, kw, sz, sy, sx, pz, py, px);
}

static void run_sbb(hipStream_t st, WS& ws, const SBBp& s, const float* x,
                    float* outb, float* tmp, Dims d, int C, const float* mask)
{
  conv_launch(st, ws, x, d, C, s.w1, C, 3, 3, 3, 1, 1, 1, 1, 1, 1, 1,
              s.bn1, s.b1, nullptr, mask, 1, tmp, d);
  conv_launch(st, ws, tmp, d, C, s.w2, C, 3, 3, 3, 1, 1, 1, 1, 1, 1, 1,
              s.bn2, s.b2, x, mask, 2, outb, d);
}

static float* run_sed(hipStream_t st, WS& ws, const SEDp& sp,
                      const float* x, const float* mask, Dims d, int C)
{
  const size_t v0 = vox(d);
  float* t1 = wsAlloc(ws, v0 * C);
  float* tA = wsAlloc(ws, v0 * C);
  float* f0 = wsAlloc(ws, v0 * C);
  // level 0 (stride 1, no 'first'): 2 SBBs
  run_sbb(st, ws, sp.sbbs[0][0], x,  tA, t1, d, C, mask);
  run_sbb(st, ws, sp.sbbs[0][1], tA, f0, t1, d, C, mask);

  // level 1: stride-2 pab_spconv + 1 SBB
  Dims d1{cdim(d.D, 3, 2, 1), cdim(d.H, 3, 2, 1), cdim(d.W, 3, 2, 1)};
  const size_t v1 = vox(d1);
  float* m1 = wsAlloc(ws, v1);
  dilate_launch(st, mask, d, m1, d1, 3, 3, 3, 2, 2, 2, 1, 1, 1);
  float* x1 = wsAlloc(ws, v1 * C);
  conv_launch(st, ws, f0, d, C, sp.first[1].w, C, 3, 3, 3, 2, 2, 2, 1, 1, 1, 1,
              sp.first[1].bn, nullptr, nullptr, m1, 1, x1, d1);
  float* u1 = wsAlloc(ws, v1 * C);
  float* f1 = wsAlloc(ws, v1 * C);
  run_sbb(st, ws, sp.sbbs[1][0], x1, f1, u1, d1, C, m1);

  // level 2
  Dims d2{cdim(d1.D, 3, 2, 1), cdim(d1.H, 3, 2, 1), cdim(d1.W, 3, 2, 1)};
  const size_t v2 = vox(d2);
  float* m2 = wsAlloc(ws, v2);
  dilate_launch(st, m1, d1, m2, d2, 3, 3, 3, 2, 2, 2, 1, 1, 1);
  float* x2 = wsAlloc(ws, v2 * C);
  conv_launch(st, ws, f1, d1, C, sp.first[2].w, C, 3, 3, 3, 2, 2, 2, 1, 1, 1, 1,
              sp.first[2].bn, nullptr, nullptr, m2, 1, x2, d2);
  float* u2 = wsAlloc(ws, v2 * C);
  float* f2 = wsAlloc(ws, v2 * C);
  run_sbb(st, ws, sp.sbbs[2][0], x2, f2, u2, d2, C, m2);

  // decoder stage 0: f2 (d2) -> d1 ; transposed conv (lhs dilation 2, padlo=1)
  float* y1 = wsAlloc(ws, v1 * C);
  conv_launch(st, ws, f2, d2, C, sp.dec[0].w, C, 3, 3, 3, 1, 1, 1, 1, 1, 1, 2,
              sp.dec[0].bn, nullptr, nullptr, m1, 1, y1, d1);
  float* z1 = wsAlloc(ws, v1 * C);
  {
    long long total = (long long)v1 * C;
    bn_add_mask_kernel<<<(int)((total + 255) / 256), 256, 0, st>>>(
        y1, f1, sp.dec[0].norm.g, sp.dec[0].norm.b, sp.dec[0].norm.m,
        sp.dec[0].norm.v, m1, z1, total, C);
  }
  // decoder stage 1: z1 (d1) -> d
  float* y0 = wsAlloc(ws, v0 * C);
  conv_launch(st, ws, z1, d1, C, sp.dec[1].w, C, 3, 3, 3, 1, 1, 1, 1, 1, 1, 2,
              sp.dec[1].bn, nullptr, nullptr, mask, 1, y0, d);
  float* z0 = wsAlloc(ws, v0 * C);
  {
    long long total = (long long)v0 * C;
    bn_add_mask_kernel<<<(int)((total + 255) / 256), 256, 0, st>>>(
        y0, f0, sp.dec[1].norm.g, sp.dec[1].norm.b, sp.dec[1].norm.m,
        sp.dec[1].norm.v, mask, z0, total, C);
  }
  return z0;
}

} // namespace

extern "C" void kernel_launch(void* const* d_in, const int* in_sizes, int n_in,
                              void* d_out, int out_size, void* d_ws, size_t ws_size,
                              hipStream_t stream)
{
  if (n_in < 4) return;
  const float* vf    = (const float*)d_in[0];
  const int*   coors = (const int*)d_in[1];
  const int    Nvox  = in_sizes[1] / 4;

  // ---- parameter pytree (sorted-dict flatten order) ----
  Cursor cur{d_in, 3};
  PABp c1_down = rdPAB(cur);
  SBBp c1_s1   = rdSBB(cur);
  SBBp c1_s2   = rdSBB(cur);
  PABp c1_stem = rdPAB(cur);
  PABp c2_down = rdPAB(cur);
  SEDp c2_sed  = rdSED(cur);
  PABp c3_down = rdPAB(cur);
  SEDp c3_sed  = rdSED(cur);
  SEDp L0 = rdSED(cur);
  SEDp L1 = rdSED(cur);
  PABp o1 = rdPAB(cur);
  PABp o2 = rdPAB(cur);

  WS ws{(char*)d_ws, 0, ws_size};

  // ---- dense scatter (NDHWC, C=5) + mask ----
  Dims d0{41, 128, 128};
  size_t v0 = vox(d0);
  float* dense5 = wsAlloc(ws, v0 * 5);
  float* mask0  = wsAlloc(ws, v0);
  (void)hipMemsetAsync(dense5, 0, v0 * 5 * sizeof(float), stream);
  (void)hipMemsetAsync(mask0, 0, v0 * sizeof(float), stream);
  scatter_kernel<<<(Nvox + 255) / 256, 256, 0, stream>>>(
      vf, coors, dense5, mask0, Nvox, d0.D, d0.H, d0.W, 5);

  // ---- conv1: stem (5->16), 2 SBBs, downsample (16->32, s2) ----
  float* A0 = wsAlloc(ws, v0 * 16);
  float* B0 = wsAlloc(ws, v0 * 16);
  float* T0 = wsAlloc(ws, v0 * 16);
  conv_launch(stream, ws, dense5, d0, 5, c1_stem.w, 16, 3, 3, 3, 1, 1, 1,
              1, 1, 1, 1, c1_stem.bn, nullptr, nullptr, mask0, 1, A0, d0);
  run_sbb(stream, ws, c1_s1, A0, B0, T0, d0, 16, mask0);
  run_sbb(stream, ws, c1_s2, B0, A0, T0, d0, 16, mask0);

  Dims d1{cdim(d0.D, 3, 2, 1), cdim(d0.H, 3, 2, 1), cdim(d0.W, 3, 2, 1)}; // 21,64,64
  size_t v1 = vox(d1);
  float* mask1 = wsAlloc(ws, v1);
  dilate_launch(stream, mask0, d0, mask1, d1, 3, 3, 3, 2, 2, 2, 1, 1, 1);
  float* X1 = wsAlloc(ws, v1 * 32);
  conv_launch(stream, ws, A0, d0, 16, c1_down.w, 32, 3, 3, 3, 2, 2, 2,
              1, 1, 1, 1, c1_down.bn, nullptr, nullptr, mask1, 1, X1, d1);

  // ---- conv2: SED(32) + downsample (32->64, s2) ----
  float* S1 = run_sed(stream, ws, c2_sed, X1, mask1, d1, 32);
  Dims d2{cdim(d1.D, 3, 2, 1), cdim(d1.H, 3, 2, 1), cdim(d1.W, 3, 2, 1)}; // 11,32,32
  size_t v2 = vox(d2);
  float* mask2 = wsAlloc(ws, v2);
  dilate_launch(stream, mask1, d1, mask2, d2, 3, 3, 3, 2, 2, 2, 1, 1, 1);
  float* X2 = wsAlloc(ws, v2 * 64);
  conv_launch(stream, ws, S1, d1, 32, c2_down.w, 64, 3, 3, 3, 2, 2, 2,
              1, 1, 1, 1, c2_down.bn, nullptr, nullptr, mask2, 1, X2, d2);

  // ---- conv3: SED(64) + downsample (64->128, s=(1,2,2)) ----
  float* S2 = run_sed(stream, ws, c3_sed, X2, mask2, d2, 64);
  Dims d3{cdim(d2.D, 3, 1, 1), cdim(d2.H, 3, 2, 1), cdim(d2.W, 3, 2, 1)}; // 11,16,16
  size_t v3 = vox(d3);
  float* mask3 = wsAlloc(ws, v3);
  dilate_launch(stream, mask2, d2, mask3, d3, 3, 3, 3, 1, 2, 2, 1, 1, 1);
  float* X3 = wsAlloc(ws, v3 * 128);
  conv_launch(stream, ws, S2, d2, 64, c3_down.w, 128, 3, 3, 3, 1, 2, 2,
              1, 1, 1, 1, c3_down.bn, nullptr, nullptr, mask3, 1, X3, d3);

  // ---- two SED(128) layers ----
  float* S3 = run_sed(stream, ws, L0, X3, mask3, d3, 128);
  float* S4 = run_sed(stream, ws, L1, S3, mask3, d3, 128);

  // ---- out1 / out2: (3,1,1) convs stride (2,1,1), pad 0 ----
  Dims d4{cdim(d3.D, 3, 2, 0), d3.H, d3.W};  // 5,16,16
  size_t v4 = vox(d4);
  float* mask4 = wsAlloc(ws, v4);
  dilate_launch(stream, mask3, d3, mask4, d4, 3, 1, 1, 2, 1, 1, 0, 0, 0);
  float* X4 = wsAlloc(ws, v4 * 128);
  conv_launch(stream, ws, S4, d3, 128, o1.w, 128, 3, 1, 1, 2, 1, 1,
              0, 0, 0, 1, o1.bn, nullptr, nullptr, mask4, 1, X4, d4);

  Dims d5{cdim(d4.D, 3, 2, 0), d4.H, d4.W};  // 2,16,16
  size_t v5 = vox(d5);
  float* mask5 = wsAlloc(ws, v5);
  dilate_launch(stream, mask4, d4, mask5, d5, 3, 1, 1, 2, 1, 1, 0, 0, 0);
  float* X5 = wsAlloc(ws, v5 * 128);
  conv_launch(stream, ws, X4, d4, 128, o2.w, 128, 3, 1, 1, 2, 1, 1,
              0, 0, 0, 1, o2.bn, nullptr, nullptr, mask5, 1, X5, d5);

  // ---- NDHWC -> NCDHW into d_out ----
  int total = 128 * d5.D * d5.H * d5.W;
  to_ncdhw_kernel<<<(total + 255) / 256, 256, 0, stream>>>(
      X5, (float*)d_out, 128, d5.D, d5.H, d5.W);
}
